// Network_85409719648558
// MI455X (gfx1250) — matrix operand order repaired and checked by reading, not compiled, not run on hardware
//
#include <hip/hip_runtime.h>
#include <hip/hip_bf16.h>

typedef _Float16 v16h __attribute__((ext_vector_type(16)));
typedef _Float16 v8h  __attribute__((ext_vector_type(8)));
typedef float    v8f  __attribute__((ext_vector_type(8)));

#define NRAYS 2048
#define NC    64
#define NNEW  128
#define NF    192
#define PERNET 593920   // packed halves per network
#define TM    32        // samples per workgroup tile

// workspace byte offsets
#define WS_PACKED 0u
#define WS_ZC   2375680u
#define WS_ZF   2899968u
#define WS_RAWC 4472832u
#define WS_RAWF 6569984u

// output float offsets: rgb_map, acc0, rgb0, z_std
#define OUT_ACC0 6144
#define OUT_RGB0 8192
#define OUT_ZSTD 14336

// ---------------- layer tables (10 WMMA layers: full0..7, feature, view) ---
__device__ __constant__ int d_WOFF[11] = {0,16384,81920,147456,212992,278528,
                                          360448,425984,491520,557056,593920};
__device__ __constant__ int d_KT[10]   = {2,8,8,8,8,10,8,8,8,9};
__device__ __constant__ int d_NT[10]   = {16,16,16,16,16,16,16,16,16,8};

struct MlpNet {
  const _Float16* packed;
  const float* bias[10];      // full0..7 bias, feature bias, view bias
  const float* alphaW; const float* alphaB;
  const float* rgbW;   const float* rgbB;
};
struct PackSrc { const float* W[10]; }; // full0..7, feature, view

// ---------------- helpers --------------------------------------------------
__device__ inline float enc_ch(const float* P, int c, int L) {
  if (c < 3) return P[c];
  int idx = c - 3;
  if (idx >= 6 * L) return 0.f;
  int l = idx / 6, rr = idx % 6;
  float f = (float)(1 << l);
  return (rr < 3) ? __sinf(P[rr] * f) : __cosf(P[rr - 3] * f);
}

// A-fragment for v_wmma_f32_16x16x32_f16 from an LDS row-major f16 tile.
// lanes 0-15: row=lane, halves 0-7 -> K=k0..k0+7,  halves 8-15 -> K=k0+16..23
// lanes16-31: row=lane-16, halves 0-7 -> K=k0+8..15, halves 8-15 -> K=k0+24..31
__device__ inline v16h load_a_frag(const _Float16* base, int stride, int k0,
                                   int laneLo, int laneHi) {
  const _Float16* p = base + laneLo * stride + k0 + laneHi * 8;
  v8h a0 = *(const v8h*)(p);
  v8h a1 = *(const v8h*)(p + 16);
  return __builtin_shufflevector(a0, a1, 0,1,2,3,4,5,6,7,8,9,10,11,12,13,14,15);
}

// ---------------- weight packing ------------------------------------------
// Fragment layout (512 halves): half index = lane*16 + h,
// lanes 0-15: col n = nt*16+lane, K = kt*32 + (h<8 ? h : 16+h-8)
// lanes16-31: col n = nt*16+lane-16, K = kt*32 + 8 + (h<8 ? h : 16+h-8)
__global__ __launch_bounds__(256) void pack_kernel(PackSrc a, PackSrc b,
                                                   _Float16* packed) {
  int gid = blockIdx.x * 256 + threadIdx.x;
  if (gid >= 2 * PERNET) return;
  int net = gid / PERNET, p = gid % PERNET;
  const PackSrc& src = net ? b : a;
  int L = 0;
  while (p >= d_WOFF[L + 1]) ++L;
  int within = p - d_WOFF[L];
  int frag = within >> 9, q = within & 511;
  int lane = q >> 4, h = q & 15;
  int nnt = d_NT[L];
  int kt = frag / nnt, nt = frag % nnt;
  int dout = (L == 9) ? 128 : 256;
  int n = nt * 16 + (lane & 15);
  int k = kt * 32 + (h & 7) + ((h >> 3) ? 16 : 0) + ((lane >> 4) ? 8 : 0);
  int r = k; bool valid = true;
  if (L == 0)      { valid = (k < 63); }
  else if (L == 5) { if (k < 63) r = k; else if (k == 63) valid = false; else r = k - 1; }
  else if (L == 9) { valid = (k < 283); }
  float v = valid ? src.W[L][r * dout + n] : 0.f;
  packed[(size_t)net * PERNET + p] = (_Float16)v;
}

// ---------------- coarse z -------------------------------------------------
__device__ inline float zbase(int i) { return 2.f + 4.f * ((float)i / 63.f); }
__global__ __launch_bounds__(256) void zc_kernel(const float* __restrict__ t_rand,
                                                 float* __restrict__ zc) {
  int g = blockIdx.x * 256 + threadIdx.x;           // NRAYS*NC threads
  int i = g & 63;
  float lo = (i == 0)  ? zbase(0)  : 0.5f * (zbase(i - 1) + zbase(i));
  float up = (i == 63) ? zbase(63) : 0.5f * (zbase(i) + zbase(i + 1));
  zc[g] = lo + (up - lo) * t_rand[g];
}

// ---------------- WMMA MLP (32-sample tiles, 8 waves) -----------------------
__global__ __launch_bounds__(256) void mlp_kernel(MlpNet net,
    const float* __restrict__ rays_o, const float* __restrict__ rays_d,
    const float* __restrict__ zbuf, float* __restrict__ raw, int S) {
  __shared__ __align__(16) _Float16 sPE[TM * 64];
  __shared__ __align__(16) _Float16 sVE[TM * 32];
  __shared__ __align__(16) _Float16 sA0[TM * 320];
  __shared__ __align__(16) _Float16 sA1[TM * 320];

  const int tid = threadIdx.x;
  const int base = blockIdx.x * TM;
  {   // encodings: thread (s,slot) fills 8 PE + 4 VE channels of sample s
    int s = tid >> 3, slot = tid & 7;
    int gs = base + s, ray = gs / S;
    float z = zbuf[gs];
    float ox = rays_o[ray*3+0], oy = rays_o[ray*3+1], oz = rays_o[ray*3+2];
    float dx = rays_d[ray*3+0], dy = rays_d[ray*3+1], dz = rays_d[ray*3+2];
    float P[3] = { ox + dx*z, oy + dy*z, oz + dz*z };
    float dn = rsqrtf(dx*dx + dy*dy + dz*dz);
    float V[3] = { dx*dn, dy*dn, dz*dn };
    for (int q = 0; q < 8; ++q) { int c = slot*8+q; sPE[s*64+c] = (_Float16)enc_ch(P, c, 10); }
    for (int q = 0; q < 4; ++q) { int c = slot*4+q; sVE[s*32+c] = (_Float16)enc_ch(V, c, 4); }
  }
  __syncthreads();

  const int lane = tid & 31, wv = tid >> 5;
  const int laneLo = lane & 15, laneHi = lane >> 4;
  float alphaVal = 0.f;

  for (int L = 0; L < 10; ++L) {
    // alpha head reads layer-7 output (sA1) before view layer overwrites it
    if (L == 8 && tid < TM) {
      const _Float16* rp = &sA1[tid * 320];
      float a = net.alphaB[0];
      for (int k = 0; k < 256; ++k) a += (float)rp[k] * net.alphaW[k];
      alphaVal = a;
    }
    // ping-pong select without pointer arrays (LDS addrspace-safe)
    const _Float16* actSrc = (L == 0) ? sA0 : (((L - 1) & 1) ? sA1 : sA0);
    _Float16* dst = (L & 1) ? sA1 : sA0;
    const int nkt = d_KT[L], nnt = d_NT[L];
    const int nt0 = wv, nt1 = wv + 8;
    v8f c00 = {0.f,0.f,0.f,0.f,0.f,0.f,0.f,0.f};
    v8f c01 = {0.f,0.f,0.f,0.f,0.f,0.f,0.f,0.f};
    v8f c10 = {0.f,0.f,0.f,0.f,0.f,0.f,0.f,0.f};
    v8f c11 = {0.f,0.f,0.f,0.f,0.f,0.f,0.f,0.f};
    const _Float16* wl = net.packed + d_WOFF[L];
    for (int kt = 0; kt < nkt; ++kt) {
      const _Float16* abase; int astride, k0;
      if (L == 0)                 { abase = sPE; astride = 64;  k0 = kt * 32; }
      else if (L == 5 && kt < 2)  { abase = sPE; astride = 64;  k0 = kt * 32; }
      else if (L == 9 && kt == 8) { abase = sVE; astride = 32;  k0 = 0; }
      else { abase = actSrc; astride = 320; k0 = ((L == 5) ? (kt - 2) : kt) * 32; }
      v16h a0 = load_a_frag(abase, astride, k0, laneLo, laneHi);           // rows 0..15
      v16h a1 = load_a_frag(abase + 16 * astride, astride, k0, laneLo, laneHi); // rows 16..31
      const _Float16* wk = wl + (size_t)(kt * nnt) * 512;
      v16h b0 = *(const v16h*)(wk + (size_t)nt0 * 512 + lane * 16);
      c00 = __builtin_amdgcn_wmma_f32_16x16x32_f16(false, a0, false, b0,
                                                   (short)0, c00, false, false);
      c10 = __builtin_amdgcn_wmma_f32_16x16x32_f16(false, a1, false, b0,
                                                   (short)0, c10, false, false);
      if (nnt == 16) {
        v16h b1 = *(const v16h*)(wk + (size_t)nt1 * 512 + lane * 16);
        c01 = __builtin_amdgcn_wmma_f32_16x16x32_f16(false, a0, false, b1,
                                                     (short)0, c01, false, false);
        c11 = __builtin_amdgcn_wmma_f32_16x16x32_f16(false, a1, false, b1,
                                                     (short)0, c11, false, false);
      }
    }
    const float* bi = net.bias[L];
    const bool doRelu = (L != 8);
    {
      float bv = bi[nt0 * 16 + laneLo];
      for (int j = 0; j < 8; ++j) {
        float v0 = c00[j] + bv, v1 = c10[j] + bv;
        if (doRelu) { v0 = fmaxf(v0, 0.f); v1 = fmaxf(v1, 0.f); }
        dst[(j + 8 * laneHi) * 320 + nt0 * 16 + laneLo]        = (_Float16)v0;
        dst[(j + 8 * laneHi + 16) * 320 + nt0 * 16 + laneLo]   = (_Float16)v1;
      }
    }
    if (nnt == 16) {
      float bv = bi[nt1 * 16 + laneLo];
      for (int j = 0; j < 8; ++j) {
        float v0 = c01[j] + bv, v1 = c11[j] + bv;
        if (doRelu) { v0 = fmaxf(v0, 0.f); v1 = fmaxf(v1, 0.f); }
        dst[(j + 8 * laneHi) * 320 + nt1 * 16 + laneLo]        = (_Float16)v0;
        dst[(j + 8 * laneHi + 16) * 320 + nt1 * 16 + laneLo]   = (_Float16)v1;
      }
    }
    __syncthreads();
  }
  // heads: view output (relu'd h, 128 cols) is in sA1 cols 0..127
  if (tid < 3 * TM) {
    int ss = tid / 3, c = tid % 3;
    const _Float16* hp = &sA1[ss * 320];
    float acc = net.rgbB[c];
    for (int k = 0; k < 128; ++k) acc += (float)hp[k] * net.rgbW[k * 3 + c];
    raw[(size_t)(base + ss) * 4 + c] = acc;
  }
  if (tid < TM) raw[(size_t)(base + tid) * 4 + 3] = alphaVal;
}

// ---------------- coarse composite + hierarchical sampling -----------------
__device__ inline float sigm(float x) { return 1.f / (1.f + __expf(-x)); }

__global__ __launch_bounds__(256) void composite_coarse_kernel(
    const float* __restrict__ rays_d, const float* __restrict__ u_samples,
    const float* __restrict__ zc, const float* __restrict__ rawc,
    float* __restrict__ zf, float* __restrict__ out) {
  __shared__ float sz[NC], sAl[NC], sW[NC], sCdf[63], sMids[63], sAll[NF];
  const int t = threadIdx.x, r = blockIdx.x;
  float dx = rays_d[r*3+0], dy = rays_d[r*3+1], dz = rays_d[r*3+2];
  float dn = sqrtf(dx*dx + dy*dy + dz*dz);
  if (t < NC) sz[t] = zc[r * NC + t];
  __syncthreads();
  if (t < NC) {
    float dist = ((t < NC - 1) ? (sz[t + 1] - sz[t]) : 1e10f) * dn;
    float sig = fmaxf(rawc[(size_t)(r * NC + t) * 4 + 3], 0.f);
    sAl[t] = 1.f - __expf(-sig * dist);
  }
  __syncthreads();
  if (t == 0) {
    float T = 1.f, acc = 0.f, rg[3] = {0.f, 0.f, 0.f};
    for (int i = 0; i < NC; ++i) {
      float w = sAl[i] * T; sW[i] = w; T *= (1.f - sAl[i] + 1e-10f);
      acc += w;
      for (int c = 0; c < 3; ++c)
        rg[c] += w * sigm(rawc[(size_t)(r * NC + i) * 4 + c]);
    }
    out[OUT_ACC0 + r] = acc;
    for (int c = 0; c < 3; ++c) out[OUT_RGB0 + r * 3 + c] = rg[c];
    for (int i = 0; i < 63; ++i) sMids[i] = 0.5f * (sz[i] + sz[i + 1]);
    float wsum = 0.f;
    for (int i = 0; i < 62; ++i) wsum += sW[i + 1] + 1e-5f;
    sCdf[0] = 0.f; float run = 0.f;
    for (int i = 0; i < 62; ++i) { run += (sW[i + 1] + 1e-5f) / wsum; sCdf[i + 1] = run; }
  }
  __syncthreads();
  if (t < NNEW) {
    float u = u_samples[r * NNEW + t];
    int idx = 0;
    for (int j = 0; j < 63; ++j) idx += (sCdf[j] <= u) ? 1 : 0;   // searchsorted right
    int lo = idx - 1; if (lo < 0) lo = 0;
    int up = idx; if (up > 62) up = 62;
    float cl = sCdf[lo], cu = sCdf[up], ml = sMids[lo], mu = sMids[up];
    float den = cu - cl; if (den < 1e-5f) den = 1.f;
    sAll[NC + t] = ml + ((u - cl) / den) * (mu - ml);
  }
  if (t < NC) sAll[t] = sz[t];
  __syncthreads();
  if (t == 0) {
    float m = 0.f;
    for (int i = 0; i < NNEW; ++i) m += sAll[NC + i];
    m /= (float)NNEW;
    float v = 0.f;
    for (int i = 0; i < NNEW; ++i) { float d = sAll[NC + i] - m; v += d * d; }
    out[OUT_ZSTD + r] = sqrtf(v / (float)NNEW);
  }
  if (t < NF) {   // rank sort -> sorted z_fine
    float v = sAll[t];
    int rank = 0;
    for (int j = 0; j < NF; ++j) {
      float vj = sAll[j];
      rank += (vj < v || (vj == v && j < t)) ? 1 : 0;
    }
    zf[r * NF + rank] = v;
  }
}

// ---------------- fine composite -------------------------------------------
__global__ __launch_bounds__(256) void composite_fine_kernel(
    const float* __restrict__ rays_d, const float* __restrict__ zf,
    const float* __restrict__ rawf, float* __restrict__ out) {
  __shared__ float sz[NF], sAl[NF];
  const int t = threadIdx.x, r = blockIdx.x;
  float dx = rays_d[r*3+0], dy = rays_d[r*3+1], dz = rays_d[r*3+2];
  float dn = sqrtf(dx*dx + dy*dy + dz*dz);
  if (t < NF) sz[t] = zf[r * NF + t];
  __syncthreads();
  if (t < NF) {
    float dist = ((t < NF - 1) ? (sz[t + 1] - sz[t]) : 1e10f) * dn;
    float sig = fmaxf(rawf[(size_t)(r * NF + t) * 4 + 3], 0.f);
    sAl[t] = 1.f - __expf(-sig * dist);
  }
  __syncthreads();
  if (t == 0) {
    float T = 1.f, rg[3] = {0.f, 0.f, 0.f};
    for (int i = 0; i < NF; ++i) {
      float w = sAl[i] * T; T *= (1.f - sAl[i] + 1e-10f);
      for (int c = 0; c < 3; ++c)
        rg[c] += w * sigm(rawf[(size_t)(r * NF + i) * 4 + c]);
    }
    for (int c = 0; c < 3; ++c) out[r * 3 + c] = rg[c];
  }
}

// ---------------- host ------------------------------------------------------
extern "C" void kernel_launch(void* const* d_in, const int* in_sizes, int n_in,
                              void* d_out, int out_size, void* d_ws, size_t ws_size,
                              hipStream_t stream) {
  (void)in_sizes; (void)n_in; (void)out_size; (void)ws_size;
  auto F = [&](int i) { return (const float*)d_in[i]; };
  const float* rays_o = F(0);
  const float* rays_d = F(1);
  const float* t_rand = F(2);
  const float* u_samp = F(3);

  char* ws = (char*)d_ws;
  _Float16* packed = (_Float16*)(ws + WS_PACKED);
  float* zc   = (float*)(ws + WS_ZC);
  float* zf   = (float*)(ws + WS_ZF);
  float* rawc = (float*)(ws + WS_RAWC);
  float* rawf = (float*)(ws + WS_RAWF);
  float* out  = (float*)d_out;

  // d_in leaf order per network (JAX dict sorted keys):
  // alphaW, alphaB, featW, featB, full0W,full0B ... full7W,full7B, rgbW, rgbB, viewW, viewB
  auto makePack = [&](int o) {
    PackSrc p;
    for (int i = 0; i < 8; ++i) p.W[i] = F(o + 4 + 2 * i);
    p.W[8] = F(o + 2); p.W[9] = F(o + 22);
    return p;
  };
  auto makeNet = [&](int o, int netIdx) {
    MlpNet n;
    n.packed = packed + (size_t)netIdx * PERNET;
    for (int i = 0; i < 8; ++i) n.bias[i] = F(o + 5 + 2 * i);
    n.bias[8] = F(o + 3); n.bias[9] = F(o + 23);
    n.alphaW = F(o + 0); n.alphaB = F(o + 1);
    n.rgbW = F(o + 20);  n.rgbB = F(o + 21);
    return n;
  };
  PackSrc pc = makePack(4), pf = makePack(28);
  MlpNet nc = makeNet(4, 0), nf = makeNet(28, 1);

  pack_kernel<<<(2 * PERNET) / 256, 256, 0, stream>>>(pc, pf, packed);
  zc_kernel<<<(NRAYS * NC) / 256, 256, 0, stream>>>(t_rand, zc);
  mlp_kernel<<<(NRAYS * NC) / TM, 256, 0, stream>>>(nc, rays_o, rays_d, zc, rawc, NC);
  composite_coarse_kernel<<<NRAYS, 256, 0, stream>>>(rays_d, u_samp, zc, rawc, zf, out);
  mlp_kernel<<<(NRAYS * NF) / TM, 256, 0, stream>>>(nf, rays_o, rays_d, zf, rawf, NF);
  composite_fine_kernel<<<NRAYS, 256, 0, stream>>>(rays_d, zf, rawf, out);
}
